// PointerNetwork_42339787604668
// MI455X (gfx1250) — matrix operand order repaired and checked by reading, not compiled, not run on hardware
//
#include <hip/hip_runtime.h>
#include <hip/hip_bf16.h>
#include <math.h>

// ---------------------------------------------------------------------------
// Pointer network for MI455X (gfx1250, wave32, WMMA).
//   B=128, L=256, IN=32, E=256, H=512, 4H=2048
// All GEMMs use v_wmma_f32_16x16x32_bf16 (16x16 tiles, fp32 accumulate).
// Weights are packed once into the WMMA B-fragment layout so the per-step
// recurrent GEMMs do a single contiguous 32B load per lane per k-tile,
// hitting L2 (~11MB of packed weights << 192MB L2).
// ---------------------------------------------------------------------------

#define BDIM 128
#define LDIM 256
#define INDIM 32
#define EDIM 256
#define HDIM 512
#define G4H  2048
#define BL   (BDIM * LDIM)

typedef __bf16 bf16_t;
typedef __attribute__((ext_vector_type(16))) __bf16 v16bf;
typedef __attribute__((ext_vector_type(8)))  float  v8f;

static __device__ __forceinline__ float bf2f(bf16_t v) {
  unsigned short u = __builtin_bit_cast(unsigned short, v);
  unsigned int x = ((unsigned int)u) << 16;
  return __builtin_bit_cast(float, x);
}
static __device__ __forceinline__ bf16_t f2bf(float f) {
  unsigned int x = __builtin_bit_cast(unsigned int, f);
  unsigned int r = x + 0x7FFFu + ((x >> 16) & 1u);   // round-to-nearest-even
  unsigned short u = (unsigned short)(r >> 16);
  return __builtin_bit_cast(bf16_t, u);
}
static __device__ __forceinline__ float sigmoidf_(float x) {
  return 1.0f / (1.0f + expf(-x));
}

// D = A(16x32) * B(32x16) + C, bf16 in / f32 out, one wave.
static __device__ __forceinline__ v8f wmma_bf16(v16bf a, v16bf b, v8f c) {
  return __builtin_amdgcn_wmma_f32_16x16x32_bf16(
      false, a, false, b, (short)0, c, false, false);
}

// A-fragment (16x32 bf16). Caller supplies per-lane pointer
//   p = rowbase(row = lane&15) + khalf   with khalf = (lane>>4)*8.
// Lane holds K = {khalf..khalf+7, khalf+16..khalf+23}  (two 16B loads).
static __device__ __forceinline__ v16bf load_a_from(const bf16_t* p) {
  v16bf a;
#pragma unroll
  for (int i = 0; i < 8; ++i) a[i] = p[i];
#pragma unroll
  for (int i = 0; i < 8; ++i) a[8 + i] = p[16 + i];
  return a;
}

// B-fragment from packed layout: lane's 16 bf16 are contiguous (32B load).
static __device__ __forceinline__ v16bf load_b_packed(const bf16_t* p) {
  v16bf b;
#pragma unroll
  for (int i = 0; i < 16; ++i) b[i] = p[i];
  return b;
}

// ---------------------------------------------------------------------------
// Utility kernels
// ---------------------------------------------------------------------------
__global__ void zero_kernel(unsigned int* p, long n) {
  long i = (long)blockIdx.x * blockDim.x + threadIdx.x;
  if (i < n) p[i] = 0u;
}
__global__ void copy_f32_kernel(const float* __restrict__ s, float* __restrict__ d, long n) {
  long i = (long)blockIdx.x * blockDim.x + threadIdx.x;
  if (i < n) d[i] = s[i];
}
__global__ void f32_to_bf16_kernel(const float* __restrict__ s, bf16_t* __restrict__ d, long n) {
  long i = (long)blockIdx.x * blockDim.x + threadIdx.x;
  if (i < n) d[i] = f2bf(s[i]);
}

// Pack B(k,n) = W^T into WMMA B-fragment layout. Source rows are row-major
// (N x Krow). K is the concatenation of W0 (k0 cols) and W1 (k1 cols).
// Storage offset == thread index: (((nt*KT + kt)*32 + lane)*16 + i)
// with  n = nt*16 + (lane&15),  k = kt*32 + (lane>>4)*8 + (i&7) + (i>>3)*16.
__global__ void pack_w_kernel(const float* __restrict__ W0, int k0,
                              const float* __restrict__ W1, int k1,
                              int KT, bf16_t* __restrict__ out, long total) {
  long idx = (long)blockIdx.x * blockDim.x + threadIdx.x;
  if (idx >= total) return;
  int i    = (int)(idx & 15);
  int lane = (int)((idx >> 4) & 31);
  long blk = idx >> 9;
  int kt = (int)(blk % KT);
  int nt = (int)(blk / KT);
  int n = nt * 16 + (lane & 15);
  int k = kt * 32 + (lane >> 4) * 8 + (i & 7) + (i >> 3) * 16;
  float v;
  if (k < k0)      v = W0[(size_t)n * k0 + k];
  else if (W1)     v = W1[(size_t)n * k1 + (k - k0)];
  else             v = 0.0f;
  out[idx] = f2bf(v);
}

// ---------------------------------------------------------------------------
// Generic WMMA GEMM: C(M x N) = A(M x K, row-major bf16) * Bpacked + bias.
// Block = 128 threads = 4 waves; wave w -> ntile = blockIdx.x*4 + w,
// mtile = blockIdx.y. Each wave owns one 16x16 f32 tile over full K.
// ---------------------------------------------------------------------------
template <bool STORE_BF16>
__global__ __launch_bounds__(128) void gemm_wmma_kernel(
    const bf16_t* __restrict__ A, int lda,
    const bf16_t* __restrict__ Bp, int KT,
    const float* __restrict__ bias,
    float* __restrict__ Cf, bf16_t* __restrict__ Cb, int ldc) {
  int lane = threadIdx.x & 31;
  int wave = threadIdx.x >> 5;
  int nt = blockIdx.x * 4 + wave;
  int mt = blockIdx.y;
  const bf16_t* arow = A + (size_t)(mt * 16 + (lane & 15)) * lda + (lane >> 4) * 8;
  const bf16_t* wp = Bp + (size_t)nt * KT * 512 + (size_t)lane * 16;
  v8f acc = {0.f, 0.f, 0.f, 0.f, 0.f, 0.f, 0.f, 0.f};
#pragma unroll 4
  for (int kt = 0; kt < KT; ++kt) {
    __builtin_prefetch((const void*)(wp + 1024), 0, 0);  // global_prefetch_b8
    v16bf a = load_a_from(arow + kt * 32);
    v16bf b = load_b_packed(wp);
    wp += 512;
    acc = wmma_bf16(a, b, acc);
  }
  int col = lane & 15;
  float bv = bias ? bias[nt * 16 + col] : 0.0f;
#pragma unroll
  for (int r = 0; r < 8; ++r) {
    int m = mt * 16 + r + 8 * (lane >> 4);
    size_t o = (size_t)m * ldc + nt * 16 + col;
    float v = acc[r] + bv;
    if (STORE_BF16) Cb[o] = f2bf(v);
    else            Cf[o] = v;
  }
}

// ---------------------------------------------------------------------------
// LSTM gates GEMM (one step):  G(B x 4H) = [x_t | h] * Wpacked + bias.
// gather=1: x rows come from emb[b, t_eff, :] with t_eff = t (fwd) or
//           clip(len[b]-1-t, 0, L-1) (bwd)  -- per-lane gather is free.
// ---------------------------------------------------------------------------
__global__ __launch_bounds__(128) void lstm_gates_kernel(
    const bf16_t* __restrict__ X, int ldx, int ktx,
    int gather, int dir, int t,
    const int* __restrict__ lens,
    const bf16_t* __restrict__ Hb,
    const bf16_t* __restrict__ Wp,
    const float* __restrict__ bias,
    float* __restrict__ G) {
  int lane = threadIdx.x & 31;
  int wave = threadIdx.x >> 5;
  int nt = blockIdx.x * 4 + wave;
  int mt = blockIdx.y;
  int b = mt * 16 + (lane & 15);
  int kh = (lane >> 4) * 8;
  const int KTH = HDIM / 32;
  const int KT = ktx + KTH;

  const bf16_t* xrow;
  if (gather) {
    int len = lens[b];
    int te;
    if (dir) {
      te = len - 1 - t;
      te = te < 0 ? 0 : (te > (LDIM - 1) ? (LDIM - 1) : te);
    } else {
      te = t;
    }
    xrow = X + ((size_t)b * LDIM + te) * ldx + kh;
  } else {
    xrow = X + (size_t)b * ldx + kh;
  }
  const bf16_t* hrow = Hb + (size_t)b * HDIM + kh;
  const bf16_t* wp = Wp + (size_t)nt * KT * 512 + (size_t)lane * 16;

  v8f acc = {0.f, 0.f, 0.f, 0.f, 0.f, 0.f, 0.f, 0.f};
#pragma unroll 4
  for (int kt = 0; kt < ktx; ++kt) {
    __builtin_prefetch((const void*)(wp + 1024), 0, 0);
    v16bf a = load_a_from(xrow + kt * 32);
    v16bf bb = load_b_packed(wp);
    wp += 512;
    acc = wmma_bf16(a, bb, acc);
  }
#pragma unroll 4
  for (int kt = 0; kt < KTH; ++kt) {
    __builtin_prefetch((const void*)(wp + 1024), 0, 0);
    v16bf a = load_a_from(hrow + kt * 32);
    v16bf bb = load_b_packed(wp);
    wp += 512;
    acc = wmma_bf16(a, bb, acc);
  }
  int col = lane & 15;
  float bv = bias[nt * 16 + col];
#pragma unroll
  for (int r = 0; r < 8; ++r) {
    int m = mt * 16 + r + 8 * (lane >> 4);
    G[(size_t)m * G4H + nt * 16 + col] = acc[r] + bv;
  }
}

// ---------------------------------------------------------------------------
// LSTM cell elementwise update. enc_mode=1: masked update + accumulate the
// (possibly time-reversed) hidden state into enc_out. enc_mode=0: decoder.
// ---------------------------------------------------------------------------
__global__ void lstm_cell_kernel(const float* __restrict__ G,
                                 float* __restrict__ h, float* __restrict__ c,
                                 bf16_t* __restrict__ hb,
                                 const int* __restrict__ lens, int t, int dir,
                                 int enc_mode, float* __restrict__ enc_out) {
  int idx = blockIdx.x * blockDim.x + threadIdx.x;
  if (idx >= BDIM * HDIM) return;
  int b = idx >> 9;       // H = 512
  int j = idx & (HDIM - 1);
  const float* g = G + (size_t)b * G4H;
  float ig = sigmoidf_(g[j]);
  float fg = sigmoidf_(g[HDIM + j]);
  float gg = tanhf(g[2 * HDIM + j]);
  float og = sigmoidf_(g[3 * HDIM + j]);
  float cn = fg * c[idx] + ig * gg;
  float hn = og * tanhf(cn);
  if (enc_mode) {
    int len = lens[b];
    bool m = t < len;
    float ho = m ? hn : h[idx];
    float co = m ? cn : c[idx];
    h[idx] = ho; c[idx] = co; hb[idx] = f2bf(ho);
    if (m) {
      int ot = dir ? (len - 1 - t) : t;
      enc_out[((size_t)b * LDIM + ot) * HDIM + j] += hn;
    }
  } else {
    h[idx] = hn; c[idx] = cn; hb[idx] = f2bf(hn);
  }
}

// ---------------------------------------------------------------------------
// Attention scores: u[b,l] = sum_j tanh(enc_proj[b,l,j] + q[b,j]) * vt[j]
// One wave32 per (b,l) row, shfl reduction.
// ---------------------------------------------------------------------------
__global__ void attn_score_kernel(const float* __restrict__ enc_proj,
                                  const float* __restrict__ q,
                                  const float* __restrict__ vt,
                                  float* __restrict__ u) {
  int gid = blockIdx.x * (blockDim.x >> 5) + (threadIdx.x >> 5);
  int lane = threadIdx.x & 31;
  if (gid >= BL) return;
  int b = gid >> 8;  // L = 256
  const float* ep = enc_proj + (size_t)gid * HDIM;
  const float* qb = q + (size_t)b * HDIM;
  float s = 0.0f;
#pragma unroll 4
  for (int j = lane; j < HDIM; j += 32) s += tanhf(ep[j] + qb[j]) * vt[j];
#pragma unroll
  for (int off = 16; off > 0; off >>= 1) s += __shfl_down(s, off, 32);
  if (lane == 0) u[gid] = s;
}

// ---------------------------------------------------------------------------
// Masked log-softmax + argmax + gather next decoder input.
// One block (256 threads == L) per batch row.
// ---------------------------------------------------------------------------
__global__ void attn_softmax_kernel(const float* __restrict__ u,
                                    const int* __restrict__ lens,
                                    const bf16_t* __restrict__ enc_bf,
                                    int i,
                                    float* __restrict__ out_logp,
                                    float* __restrict__ out_arg,
                                    bf16_t* __restrict__ x_next) {
  __shared__ float red[LDIM];
  __shared__ int redi[LDIM];
  int b = blockIdx.x;
  int tid = threadIdx.x;  // == l
  int len = lens[b];
  const float LOGEPS = -103.2789f;  // logf(float(1e-45))
  bool valid = (tid < len) && (i < len);
  float val = u[(size_t)b * LDIM + tid] + (valid ? 0.0f : LOGEPS);

  red[tid] = val; __syncthreads();
  for (int s = LDIM / 2; s > 0; s >>= 1) {
    if (tid < s) red[tid] = fmaxf(red[tid], red[tid + s]);
    __syncthreads();
  }
  float gmax = red[0]; __syncthreads();
  red[tid] = expf(val - gmax); __syncthreads();
  for (int s = LDIM / 2; s > 0; s >>= 1) {
    if (tid < s) red[tid] += red[tid + s];
    __syncthreads();
  }
  float logZ = gmax + logf(red[0]); __syncthreads();
  float logp = val - logZ;
  out_logp[(size_t)b * LDIM * LDIM + (size_t)i * LDIM + tid] = logp;

  float mval = valid ? logp : -10000000.0f;
  red[tid] = mval; redi[tid] = tid; __syncthreads();
  for (int s = LDIM / 2; s > 0; s >>= 1) {
    if (tid < s) {
      bool take = (red[tid + s] > red[tid]) ||
                  (red[tid + s] == red[tid] && redi[tid + s] < redi[tid]);
      if (take) { red[tid] = red[tid + s]; redi[tid] = redi[tid + s]; }
    }
    __syncthreads();
  }
  int am = redi[0];
  if (tid == 0) out_arg[(size_t)b * LDIM + i] = (float)am;
  for (int j = tid; j < HDIM; j += LDIM)
    x_next[(size_t)b * HDIM + j] = enc_bf[((size_t)b * LDIM + am) * HDIM + j];
}

__global__ void mask_kernel(const int* __restrict__ lens, float* __restrict__ out) {
  long idx = (long)blockIdx.x * blockDim.x + threadIdx.x;
  if (idx >= (long)BDIM * LDIM * LDIM) return;
  int b = (int)(idx / (LDIM * LDIM));
  int r = (int)((idx / LDIM) % LDIM);
  int cc = (int)(idx % LDIM);
  int len = lens[b];
  out[idx] = (r < len && cc < len) ? 1.0f : 0.0f;
}

// ---------------------------------------------------------------------------
extern "C" void kernel_launch(void* const* d_in, const int* in_sizes, int n_in,
                              void* d_out, int out_size, void* d_ws, size_t ws_size,
                              hipStream_t stream) {
  const float* input_seq = (const float*)d_in[0];
  const int*   lens      = (const int*)d_in[1];
  const float* emb_W     = (const float*)d_in[2];
  const float* Wih_f     = (const float*)d_in[3];
  const float* Whh_f     = (const float*)d_in[4];
  const float* b_f       = (const float*)d_in[5];
  const float* Wih_b     = (const float*)d_in[6];
  const float* Whh_b     = (const float*)d_in[7];
  const float* b_b       = (const float*)d_in[8];
  const float* Wd_ih     = (const float*)d_in[9];
  const float* Wd_hh     = (const float*)d_in[10];
  const float* bd        = (const float*)d_in[11];
  const float* W1        = (const float*)d_in[12];
  const float* W2        = (const float*)d_in[13];
  const float* vt        = (const float*)d_in[14];

  float* outf = (float*)d_out;
  float* out_logp = outf;                                    // (B,L,L)
  float* out_arg  = outf + (size_t)BDIM * LDIM * LDIM;       // (B,L)
  float* out_mask = out_arg + (size_t)BDIM * LDIM;           // (B,L,L)

  // ---- workspace layout (~193 MB) ----
  char* base = (char*)d_ws;
  size_t off = 0;
  auto alloc = [&](size_t bytes) -> char* {
    char* p = base + off;
    off += (bytes + 255) & ~(size_t)255;
    return p;
  };
  bf16_t* x_in_bf = (bf16_t*)alloc((size_t)BL * INDIM * 2);
  bf16_t* emb_bf  = (bf16_t*)alloc((size_t)BL * EDIM * 2);
  bf16_t* embW_p  = (bf16_t*)alloc((size_t)INDIM * EDIM * 2);
  bf16_t* Wf_p    = (bf16_t*)alloc((size_t)(EDIM + HDIM) * G4H * 2);
  bf16_t* Wb_p    = (bf16_t*)alloc((size_t)(EDIM + HDIM) * G4H * 2);
  bf16_t* Wd_p    = (bf16_t*)alloc((size_t)(2 * HDIM) * G4H * 2);
  bf16_t* W1_p    = (bf16_t*)alloc((size_t)HDIM * HDIM * 2);
  bf16_t* W2_p    = (bf16_t*)alloc((size_t)HDIM * HDIM * 2);
  float*  G       = (float*)alloc((size_t)BDIM * G4H * 4);
  float*  h_f = (float*)alloc((size_t)BDIM * HDIM * 4);
  float*  c_f = (float*)alloc((size_t)BDIM * HDIM * 4);
  float*  h_b = (float*)alloc((size_t)BDIM * HDIM * 4);
  float*  c_b = (float*)alloc((size_t)BDIM * HDIM * 4);
  float*  h_d = (float*)alloc((size_t)BDIM * HDIM * 4);
  float*  c_d = (float*)alloc((size_t)BDIM * HDIM * 4);
  bf16_t* h_f_bf = (bf16_t*)alloc((size_t)BDIM * HDIM * 2);
  bf16_t* h_b_bf = (bf16_t*)alloc((size_t)BDIM * HDIM * 2);
  bf16_t* h_d_bf = (bf16_t*)alloc((size_t)BDIM * HDIM * 2);
  bf16_t* x_d    = (bf16_t*)alloc((size_t)BDIM * HDIM * 2);
  float*  enc_out  = (float*)alloc((size_t)BL * HDIM * 4);
  bf16_t* enc_bf   = (bf16_t*)alloc((size_t)BL * HDIM * 2);
  float*  enc_proj = (float*)alloc((size_t)BL * HDIM * 4);
  float*  q        = (float*)alloc((size_t)BDIM * HDIM * 4);
  float*  u        = (float*)alloc((size_t)BL * 4);
  (void)ws_size; (void)in_sizes; (void)n_in; (void)out_size;

  const int TPB = 256;
  auto blk = [](long n, int t) { return (unsigned)((n + t - 1) / t); };

  // ---- one-time prep: convert input, pack all weights into B-fragments ----
  long nIn = (long)BL * INDIM;
  f32_to_bf16_kernel<<<blk(nIn, TPB), TPB, 0, stream>>>(input_seq, x_in_bf, nIn);
  pack_w_kernel<<<blk((long)INDIM * EDIM, TPB), TPB, 0, stream>>>(
      emb_W, INDIM, nullptr, 0, INDIM / 32, embW_p, (long)INDIM * EDIM);
  pack_w_kernel<<<blk((long)(EDIM + HDIM) * G4H, TPB), TPB, 0, stream>>>(
      Wih_f, EDIM, Whh_f, HDIM, (EDIM + HDIM) / 32, Wf_p, (long)(EDIM + HDIM) * G4H);
  pack_w_kernel<<<blk((long)(EDIM + HDIM) * G4H, TPB), TPB, 0, stream>>>(
      Wih_b, EDIM, Whh_b, HDIM, (EDIM + HDIM) / 32, Wb_p, (long)(EDIM + HDIM) * G4H);
  pack_w_kernel<<<blk((long)(2 * HDIM) * G4H, TPB), TPB, 0, stream>>>(
      Wd_ih, HDIM, Wd_hh, HDIM, (2 * HDIM) / 32, Wd_p, (long)(2 * HDIM) * G4H);
  pack_w_kernel<<<blk((long)HDIM * HDIM, TPB), TPB, 0, stream>>>(
      W1, HDIM, nullptr, 0, HDIM / 32, W1_p, (long)HDIM * HDIM);
  pack_w_kernel<<<blk((long)HDIM * HDIM, TPB), TPB, 0, stream>>>(
      W2, HDIM, nullptr, 0, HDIM / 32, W2_p, (long)HDIM * HDIM);

  // emb = input @ emb_W.T  -> bf16 (B*L x E)
  gemm_wmma_kernel<true><<<dim3(EDIM / 64, BL / 16), 128, 0, stream>>>(
      x_in_bf, INDIM, embW_p, INDIM / 32, nullptr, nullptr, emb_bf, EDIM);

  // ---- zero states / encoder accumulator ----
  auto zero_words = [&](void* p, size_t bytes) {
    long w = (long)(bytes / 4);
    zero_kernel<<<blk(w, TPB), TPB, 0, stream>>>((unsigned int*)p, w);
  };
  zero_words(h_f, (size_t)BDIM * HDIM * 4);
  zero_words(c_f, (size_t)BDIM * HDIM * 4);
  zero_words(h_b, (size_t)BDIM * HDIM * 4);
  zero_words(c_b, (size_t)BDIM * HDIM * 4);
  zero_words(h_f_bf, (size_t)BDIM * HDIM * 2);
  zero_words(h_b_bf, (size_t)BDIM * HDIM * 2);
  zero_words(x_d, (size_t)BDIM * HDIM * 2);
  zero_words(enc_out, (size_t)BL * HDIM * 4);

  // ---- bidirectional encoder: 256 steps, weight-stationary in L2 ----
  dim3 ggrid(G4H / 64, BDIM / 16);  // (32, 8) blocks of 128 threads
  unsigned cellBlocks = blk((long)BDIM * HDIM, TPB);
  for (int t = 0; t < LDIM; ++t) {
    lstm_gates_kernel<<<ggrid, 128, 0, stream>>>(
        emb_bf, EDIM, EDIM / 32, 1, 0, t, lens, h_f_bf, Wf_p, b_f, G);
    lstm_cell_kernel<<<cellBlocks, TPB, 0, stream>>>(
        G, h_f, c_f, h_f_bf, lens, t, 0, 1, enc_out);
    lstm_gates_kernel<<<ggrid, 128, 0, stream>>>(
        emb_bf, EDIM, EDIM / 32, 1, 1, t, lens, h_b_bf, Wb_p, b_b, G);
    lstm_cell_kernel<<<cellBlocks, TPB, 0, stream>>>(
        G, h_b, c_b, h_b_bf, lens, t, 1, 1, enc_out);
  }

  // enc_out -> bf16, enc_proj = enc_out @ W1.T (32768x512x512 WMMA GEMM)
  long nEnc = (long)BL * HDIM;
  f32_to_bf16_kernel<<<blk(nEnc, TPB), TPB, 0, stream>>>(enc_out, enc_bf, nEnc);
  gemm_wmma_kernel<false><<<dim3(HDIM / 64, BL / 16), 128, 0, stream>>>(
      enc_bf, HDIM, W1_p, HDIM / 32, nullptr, enc_proj, nullptr, HDIM);

  // mask output (independent)
  mask_kernel<<<blk((long)BDIM * LDIM * LDIM, TPB), TPB, 0, stream>>>(lens, out_mask);

  // ---- decoder init: (h,c) = forward-LSTM final state, x0 = 0 ----
  long nHC = (long)BDIM * HDIM;
  copy_f32_kernel<<<blk(nHC, TPB), TPB, 0, stream>>>(h_f, h_d, nHC);
  copy_f32_kernel<<<blk(nHC, TPB), TPB, 0, stream>>>(c_f, c_d, nHC);
  f32_to_bf16_kernel<<<blk(nHC, TPB), TPB, 0, stream>>>(h_f, h_d_bf, nHC);

  // ---- decoder: 256 pointer steps ----
  for (int i = 0; i < LDIM; ++i) {
    lstm_gates_kernel<<<ggrid, 128, 0, stream>>>(
        x_d, HDIM, HDIM / 32, 0, 0, i, lens, h_d_bf, Wd_p, bd, G);
    lstm_cell_kernel<<<cellBlocks, TPB, 0, stream>>>(
        G, h_d, c_d, h_d_bf, lens, i, 0, 0, nullptr);
    gemm_wmma_kernel<false><<<dim3(HDIM / 64, BDIM / 16), 128, 0, stream>>>(
        h_d_bf, HDIM, W2_p, HDIM / 32, nullptr, q, nullptr, HDIM);
    attn_score_kernel<<<BL / 8, 256, 0, stream>>>(enc_proj, q, vt, u);
    attn_softmax_kernel<<<BDIM, LDIM, 0, stream>>>(
        u, lens, enc_bf, i, out_logp, out_arg, x_d);
  }
}
